// MultiHeadAttention_5849745457395
// MI455X (gfx1250) — compile-verified, matrix operand order checked by reading, not run on hardware
//
#include <hip/hip_runtime.h>
#include <math.h>

// Problem constants (from reference)
#define BB_    4
#define LL_    1024
#define EMB_   1024
#define NH_    16
#define DH_    64
#define INNER_ 1024

// Fused-attention tiling
#define QB_    32        // query rows per block
#define LDP_   1032      // padded LDS row stride (floats) -> de-phases 64 banks

typedef __attribute__((ext_vector_type(16))) __bf16 v16bf;
typedef __attribute__((ext_vector_type(8)))  __bf16 v8bf;
typedef __attribute__((ext_vector_type(8)))  float  v8f;

// ---------------------------------------------------------------------------
// gfx1250 async global<->LDS paths (params are AS1/AS3 pointers to 16B int
// vectors -- confirmed by clang diagnostics in an earlier probe round)
// ---------------------------------------------------------------------------
#if defined(__has_builtin)
#if __has_builtin(__builtin_amdgcn_global_load_async_to_lds_b128)
#define HAVE_ASYNC_LD_LDS 1
#endif
#if __has_builtin(__builtin_amdgcn_global_store_async_from_lds_b128)
#define HAVE_ASYNC_ST_LDS 1
#endif
#if __has_builtin(__builtin_amdgcn_s_wait_asynccnt)
#define HAVE_WAIT_ASYNC 1
#endif
#endif

typedef __attribute__((__vector_size__(4 * sizeof(int)))) int v4i_;
typedef __attribute__((address_space(1))) v4i_* gas_v4i;
typedef __attribute__((address_space(3))) v4i_* las_v4i;

__device__ __forceinline__ void async_g2l_16B(const void* g, void* l) {
#ifdef HAVE_ASYNC_LD_LDS
  __builtin_amdgcn_global_load_async_to_lds_b128(
      (gas_v4i)(void*)(g), (las_v4i)(l), 0, 0);
#else
  *(uint4*)l = *(const uint4*)g;
#endif
}

__device__ __forceinline__ void async_l2g_16B(void* g, const void* l) {
#ifdef HAVE_ASYNC_ST_LDS
  __builtin_amdgcn_global_store_async_from_lds_b128(
      (gas_v4i)(g), (las_v4i)(void*)(l), 0, 0);
#else
  *(uint4*)g = *(const uint4*)l;
#endif
}

__device__ __forceinline__ void wait_async_zero() {
#ifdef HAVE_WAIT_ASYNC
  __builtin_amdgcn_s_wait_asynccnt(0);
#endif
}

// ---------------------------------------------------------------------------
// helpers
// ---------------------------------------------------------------------------
__device__ __forceinline__ unsigned short f32_to_bf16_bits(float f) {
  unsigned int u = __float_as_uint(f);
  u += 0x7fffu + ((u >> 16) & 1u);       // round-to-nearest-even
  return (unsigned short)(u >> 16);
}

// A/B fragment (16x32 bf16), row-major source with leading dim `ld`.
__device__ __forceinline__ v16bf load_frag_bf16(const unsigned short* __restrict__ base,
                                                int row, int ld, int k0, int lane) {
  const int m = lane & 15, half = lane >> 4;
  const unsigned short* p = base + (size_t)(row + m) * ld + k0 + 8 * half;
  v8bf lo = *(const v8bf*)p;
  v8bf hi = *(const v8bf*)(p + 16);
  v16bf r;
#pragma unroll
  for (int i = 0; i < 8; ++i) { r[i] = lo[i]; r[i + 8] = hi[i]; }
  return r;
}

// Same fragment, f32 source (attention probabilities in LDS), cvt in-register.
__device__ __forceinline__ v16bf load_frag_f32cvt(const float* __restrict__ base,
                                                  int row, int ld, int k0, int lane) {
  const int m = lane & 15, half = lane >> 4;
  const float* p = base + (size_t)(row + m) * ld + k0 + 8 * half;
  v16bf r;
#pragma unroll
  for (int i = 0; i < 8; ++i) {
    r[i]     = __builtin_bit_cast(__bf16, f32_to_bf16_bits(p[i]));
    r[i + 8] = __builtin_bit_cast(__bf16, f32_to_bf16_bits(p[i + 16]));
  }
  return r;
}

#define WMMA_BF16(A, Bm, C) \
  __builtin_amdgcn_wmma_f32_16x16x32_bf16(false, (A), false, (Bm), (short)0, (C), false, false)

// ---------------------------------------------------------------------------
// conversion / transpose kernels
// ---------------------------------------------------------------------------
__global__ void k_cvt_bf16(const float* __restrict__ src,
                           unsigned short* __restrict__ dst, int n) {
  int i = blockIdx.x * blockDim.x + threadIdx.x;
  if (i < n) dst[i] = f32_to_bf16_bits(src[i]);
}

// src [rows, cols] f32 -> dst [cols, rows] bf16
__global__ void k_transpose_cvt(const float* __restrict__ src,
                                unsigned short* __restrict__ dst,
                                int rows, int cols) {
  int i = blockIdx.x * blockDim.x + threadIdx.x;
  if (i < rows * cols) {
    int r = i / cols, c = i % cols;
    dst[(size_t)c * rows + r] = f32_to_bf16_bits(src[i]);
  }
}

// ---------------------------------------------------------------------------
// qkv = x @ w_qkv ; splits into Q [B,H,L,D], K [B,H,L,D], V^T [B,H,D,L] (bf16)
// wave strip 16(M)x64(N); copy-free ping-pong K pipeline
// ---------------------------------------------------------------------------
__global__ void __launch_bounds__(256, 1)
k_qkv_gemm(const unsigned short* __restrict__ xb,
           const unsigned short* __restrict__ wT,
           unsigned short* __restrict__ q,
           unsigned short* __restrict__ kk,
           unsigned short* __restrict__ vT) {
  const int lane = threadIdx.x & 31, wave = threadIdx.x >> 5;
  const int id = blockIdx.x * 8 + wave;
  const int mt = id & 255;
  const int ng = id >> 8;             // 0..47
  const int m0 = mt * 16, n0 = ng * 64;
  v8f acc[4] = {};

  v16bf a0 = load_frag_bf16(xb, m0, EMB_, 0, lane);
  v16bf a1 = load_frag_bf16(xb, m0, EMB_, 32, lane);
  v16bf b0[4], b1[4];
#pragma unroll
  for (int t = 0; t < 4; ++t) {
    b0[t] = load_frag_bf16(wT, n0 + t * 16, EMB_, 0, lane);
    b1[t] = load_frag_bf16(wT, n0 + t * 16, EMB_, 32, lane);
  }

  for (int k0 = 0; k0 < EMB_; k0 += 64) {
#pragma unroll
    for (int t = 0; t < 4; ++t) acc[t] = WMMA_BF16(a0, b0[t], acc[t]);
    if (k0 + 64 < EMB_) {
      a0 = load_frag_bf16(xb, m0, EMB_, k0 + 64, lane);
#pragma unroll
      for (int t = 0; t < 4; ++t) b0[t] = load_frag_bf16(wT, n0 + t * 16, EMB_, k0 + 64, lane);
    }
#pragma unroll
    for (int t = 0; t < 4; ++t) acc[t] = WMMA_BF16(a1, b1[t], acc[t]);
    if (k0 + 96 < EMB_) {
      a1 = load_frag_bf16(xb, m0, EMB_, k0 + 96, lane);
#pragma unroll
      for (int t = 0; t < 4; ++t) b1[t] = load_frag_bf16(wT, n0 + t * 16, EMB_, k0 + 96, lane);
    }
  }

  const int half = lane >> 4, nl = lane & 15;
#pragma unroll
  for (int t = 0; t < 4; ++t) {
    const int c = n0 + t * 16 + nl;   // column in [0, 3072)
#pragma unroll
    for (int r = 0; r < 8; ++r) {
      const int tok = m0 + r + 8 * half;           // global token [0,4096)
      const int b_ = tok >> 10, l = tok & (LL_ - 1);
      const unsigned short bv = f32_to_bf16_bits(acc[t][r]);
      if (c < INNER_) {
        const int h = c >> 6, d = c & 63;
        q[((size_t)(b_ * NH_ + h) * LL_ + l) * DH_ + d] = bv;
      } else if (c < 2 * INNER_) {
        const int cc = c - INNER_, h = cc >> 6, d = cc & 63;
        kk[((size_t)(b_ * NH_ + h) * LL_ + l) * DH_ + d] = bv;
      } else {
        const int cc = c - 2 * INNER_, h = cc >> 6, d = cc & 63;
        vT[((size_t)(b_ * NH_ + h) * DH_ + d) * LL_ + l] = bv;
      }
    }
  }
}

// ---------------------------------------------------------------------------
// FUSED attention: per block = (b,h) x 32-query strip
//   phase 0: async-stage Q strip (32x64 bf16, contiguous 4KB) to LDS
//   phase 1: S = Q K^T * scale -> LDS (32 x 1024 f32, padded rows)
//   phase 2: row softmax in LDS
//   phase 3: async-store attn block (LDS -> HBM, overlaps phase 4)
//   phase 4: ctx = P @ V (A from LDS w/ f32->bf16 cvt, B = V^T global)
// LDS: 32*1032*4 = 132096 (sP) + 4096 (sQ) + 2*1024 (partials) = 138240 B
// ---------------------------------------------------------------------------
__global__ void __launch_bounds__(256, 1)
k_fused_attn(const unsigned short* __restrict__ q,
             const unsigned short* __restrict__ kk,
             const unsigned short* __restrict__ vT,
             float* __restrict__ attn,
             unsigned short* __restrict__ ctx) {
  extern __shared__ char smem[];
  float* sP = (float*)smem;                                      // [32][LDP_]
  unsigned short* sQ = (unsigned short*)(smem + 132096);         // [32][64]
  float* sMax = (float*)(smem + 136192);                         // [32][8]
  float* sSum = (float*)(smem + 137216);                         // [32][8]

  const int tid  = threadIdx.x;
  const int lane = tid & 31, wave = tid >> 5;
  const int bh = blockIdx.x >> 5;          // [0,64)
  const int qb = blockIdx.x & 31;          // [0,32)
  const int q0 = qb * QB_;

  const unsigned short* qg = q  + (size_t)bh * LL_ * DH_ + (size_t)q0 * DH_;
  const unsigned short* kb = kk + (size_t)bh * LL_ * DH_;
  const unsigned short* vb = vT + (size_t)bh * DH_ * LL_;
  float* ag = attn + (size_t)bh * LL_ * LL_ + (size_t)q0 * LL_;  // contiguous 128KB

  // ---- phase 0: stage Q strip (32*64*2 = 4096 B, one 16B chunk per thread)
  async_g2l_16B((const char*)qg + tid * 16, (char*)sQ + tid * 16);
  wait_async_zero();
  __syncthreads();

  // ---- phase 1: scores into LDS
  {
#pragma unroll
    for (int s = 0; s < 4; ++s) {
      const int sid = s * 8 + wave;        // [0,32)
      const int mt = sid & 1;              // 2 m-tiles (16 rows each)
      const int ng = sid >> 1;             // 16 n-groups (64 keys each)
      v8f acc[4] = {};
      v16bf aA = load_frag_bf16(sQ, mt * 16, DH_, 0, lane);
      v16bf aB = load_frag_bf16(sQ, mt * 16, DH_, 32, lane);
#pragma unroll
      for (int t = 0; t < 4; ++t) {
        v16bf b = load_frag_bf16(kb, ng * 64 + t * 16, DH_, 0, lane);
        acc[t] = WMMA_BF16(aA, b, acc[t]);
      }
#pragma unroll
      for (int t = 0; t < 4; ++t) {
        v16bf b = load_frag_bf16(kb, ng * 64 + t * 16, DH_, 32, lane);
        acc[t] = WMMA_BF16(aB, b, acc[t]);
      }
      const int half = lane >> 4, nl = lane & 15;
#pragma unroll
      for (int t = 0; t < 4; ++t) {
#pragma unroll
        for (int r = 0; r < 8; ++r) {
          sP[(size_t)(mt * 16 + r + 8 * half) * LDP_ + ng * 64 + t * 16 + nl] =
              acc[t][r] * 0.125f;          // 1/sqrt(64)
        }
      }
    }
  }
  __syncthreads();

  // ---- phase 2: row softmax (8 threads per row, 128 elems each)
  {
    const int r = tid >> 3, seg = tid & 7;
    float* row = sP + (size_t)r * LDP_ + seg * 128;
    float mx = -3.4e38f;
#pragma unroll 8
    for (int i = 0; i < 128; ++i) mx = fmaxf(mx, row[i]);
    sMax[r * 8 + seg] = mx;
    __syncthreads();
    mx = sMax[r * 8];
#pragma unroll
    for (int j = 1; j < 8; ++j) mx = fmaxf(mx, sMax[r * 8 + j]);
    float sum = 0.f;
#pragma unroll 8
    for (int i = 0; i < 128; ++i) { float e = __expf(row[i] - mx); row[i] = e; sum += e; }
    sSum[r * 8 + seg] = sum;
    __syncthreads();
    sum = 0.f;
#pragma unroll
    for (int j = 0; j < 8; ++j) sum += sSum[r * 8 + j];
    const float inv = 1.0f / sum;
#pragma unroll 8
    for (int i = 0; i < 128; ++i) row[i] *= inv;
  }
  __syncthreads();

  // ---- phase 3: stream attn block out (async LDS->HBM; overlaps phase 4)
  {
    // 32 rows x 1024 f32 = 8192 float4's; 32 per thread
    for (int i = tid; i < (QB_ * LL_) / 4; i += 256) {
      const int r = i >> 8, c4 = i & 255;
      async_l2g_16B((char*)(ag + (size_t)r * LL_) + c4 * 16,
                    (const char*)(sP + (size_t)r * LDP_) + c4 * 16);
    }
  }
  // no barrier needed: phase 4 only READS sP; stores drain by s_endpgm

  // ---- phase 4: ctx = P @ V  (one 16x16 strip per wave: 2 mt x 4 nt)
  //      two independent accumulator chains to break WMMA RAW serialization
  {
    const int mt = wave >> 2;              // 0..1
    const int nt = wave & 3;               // 0..3
    v8f accE = {}, accO = {};
    for (int k0 = 0; k0 < LL_; k0 += 64) {
      v16bf aE = load_frag_f32cvt(sP, mt * 16, LDP_, k0, lane);
      v16bf bE = load_frag_bf16(vb, nt * 16, LL_, k0, lane);
      v16bf aO = load_frag_f32cvt(sP, mt * 16, LDP_, k0 + 32, lane);
      v16bf bO = load_frag_bf16(vb, nt * 16, LL_, k0 + 32, lane);
      accE = WMMA_BF16(aE, bE, accE);
      accO = WMMA_BF16(aO, bO, accO);
    }
    const int half = lane >> 4, nl = lane & 15;
    const int b_ = bh >> 4, h = bh & 15;
#pragma unroll
    for (int r = 0; r < 8; ++r) {
      const int l = q0 + mt * 16 + r + 8 * half;
      const int col = h * DH_ + nt * 16 + nl;
      ctx[(size_t)(b_ * LL_ + l) * INNER_ + col] = f32_to_bf16_bits(accE[r] + accO[r]);
    }
  }
}

// ---------------------------------------------------------------------------
// out = ctx @ w_out + b_out (f32), copy-free ping-pong K pipeline
// ---------------------------------------------------------------------------
__global__ void __launch_bounds__(256, 1)
k_out(const unsigned short* __restrict__ ctx,
      const unsigned short* __restrict__ wT,
      const float* __restrict__ bias,
      float* __restrict__ out) {
  const int lane = threadIdx.x & 31, wave = threadIdx.x >> 5;
  const int id = blockIdx.x * 8 + wave;   // [0, 4096)
  const int mt = id & 255;
  const int ng = id >> 8;                 // [0, 16)
  const int m0 = mt * 16, n0 = ng * 64;
  v8f acc[4] = {};

  v16bf a0 = load_frag_bf16(ctx, m0, INNER_, 0, lane);
  v16bf a1 = load_frag_bf16(ctx, m0, INNER_, 32, lane);
  v16bf b0[4], b1[4];
#pragma unroll
  for (int t = 0; t < 4; ++t) {
    b0[t] = load_frag_bf16(wT, n0 + t * 16, INNER_, 0, lane);
    b1[t] = load_frag_bf16(wT, n0 + t * 16, INNER_, 32, lane);
  }

  for (int k0 = 0; k0 < INNER_; k0 += 64) {
#pragma unroll
    for (int t = 0; t < 4; ++t) acc[t] = WMMA_BF16(a0, b0[t], acc[t]);
    if (k0 + 64 < INNER_) {
      a0 = load_frag_bf16(ctx, m0, INNER_, k0 + 64, lane);
#pragma unroll
      for (int t = 0; t < 4; ++t) b0[t] = load_frag_bf16(wT, n0 + t * 16, INNER_, k0 + 64, lane);
    }
#pragma unroll
    for (int t = 0; t < 4; ++t) acc[t] = WMMA_BF16(a1, b1[t], acc[t]);
    if (k0 + 96 < INNER_) {
      a1 = load_frag_bf16(ctx, m0, INNER_, k0 + 96, lane);
#pragma unroll
      for (int t = 0; t < 4; ++t) b1[t] = load_frag_bf16(wT, n0 + t * 16, INNER_, k0 + 96, lane);
    }
  }

  const int half = lane >> 4, nl = lane & 15;
#pragma unroll
  for (int t = 0; t < 4; ++t) {
    const int col = n0 + t * 16 + nl;
    const float bc = bias[col];
#pragma unroll
    for (int r = 0; r < 8; ++r) {
      out[(size_t)(m0 + r + 8 * half) * EMB_ + col] = acc[t][r] + bc;
    }
  }
}

// ---------------------------------------------------------------------------
// launch
// ---------------------------------------------------------------------------
extern "C" void kernel_launch(void* const* d_in, const int* in_sizes, int n_in,
                              void* d_out, int out_size, void* d_ws, size_t ws_size,
                              hipStream_t stream) {
  (void)in_sizes; (void)n_in; (void)out_size; (void)ws_size;
  const float* x     = (const float*)d_in[0];   // [4,1024,1024]
  const float* w_qkv = (const float*)d_in[1];   // [1024, 3072]
  const float* w_out = (const float*)d_in[2];   // [1024, 1024]
  const float* b_out = (const float*)d_in[3];   // [1024]

  float* out  = (float*)d_out;                          // [4,1024,1024]
  float* attn = out + (size_t)BB_ * LL_ * EMB_;         // [4,16,1024,1024]

  char* ws = (char*)d_ws;
  unsigned short* xb     = (unsigned short*)(ws);             //  8 MB [4096,1024]
  unsigned short* wqkvT  = (unsigned short*)(ws + 8388608);   //  6 MB [3072,1024]
  unsigned short* woutT  = (unsigned short*)(ws + 14680064);  //  2 MB [1024,1024]
  unsigned short* qbuf   = (unsigned short*)(ws + 16777216);  //  8 MB [B,H,L,D]
  unsigned short* kbuf   = (unsigned short*)(ws + 25165824);  //  8 MB [B,H,L,D]
  unsigned short* vTbuf  = (unsigned short*)(ws + 33554432);  //  8 MB [B,H,D,L]
  unsigned short* ctxbuf = (unsigned short*)(ws + 41943040);  //  8 MB [4096,1024]

  k_cvt_bf16<<<(BB_ * LL_ * EMB_) / 256, 256, 0, stream>>>(x, xb, BB_ * LL_ * EMB_);
  k_transpose_cvt<<<(EMB_ * 3 * INNER_) / 256, 256, 0, stream>>>(w_qkv, wqkvT, EMB_, 3 * INNER_);
  k_transpose_cvt<<<(INNER_ * EMB_) / 256, 256, 0, stream>>>(w_out, woutT, INNER_, EMB_);

  k_qkv_gemm<<<1536, 256, 0, stream>>>(xb, wqkvT, qbuf, kbuf, vTbuf);

  // 64 (b,h) x 32 query-strips; 138240 B dynamic LDS (2 blocks / 320KB WGP)
  k_fused_attn<<<2048, 256, 138240, stream>>>(qbuf, kbuf, vTbuf, attn, ctxbuf);

  k_out<<<512, 256, 0, stream>>>(ctxbuf, woutT, b_out, out);
}
